// HeteroGAT_50105088475755
// MI455X (gfx1250) — matrix operand order repaired and checked by reading, not compiled, not run on hardware
//
#include <hip/hip_runtime.h>
#include <math.h>

#define HD   64
#define NB   100000
#define NG   100000
#define EBB  1600000
#define EGB  200000
#define EBG  200000

typedef __attribute__((ext_vector_type(16))) _Float16 v16h;
typedef __attribute__((ext_vector_type(8)))  float    v8f;

// ---------------------------------------------------------------------------
// GEMM: H[N,64] = X[N,64] @ W[64,64], fp32 in/out, f16 WMMA with f32 accum.
// One wave per 16x16 output tile; blockDim=(32,4): 4 waves cover all 64 cols.
// A fragment (16x32 f16): lane<16 -> M=lane,   K = ko + {0..15}
//                         lane>=16-> M=lane-16,K = ko + {16..31}
// B fragment (32x16 f16): lane holds column (ct*16 + lane&15),
//                         K = ko + (lane>>4)*16 + {0..15} packed in pairs.
// D (16x16 f32, 8 VGPRs): VGPR v -> M = v + 8*(lane>>4), N = lane&15.
// ---------------------------------------------------------------------------
__global__ __launch_bounds__(128) void gemm64_wmma(
    const float* __restrict__ X, const float* __restrict__ W,
    float* __restrict__ Hout)
{
  const int lane = threadIdx.x;        // 0..31
  const int ct   = threadIdx.y;        // 0..3 column tile
  const int row0 = blockIdx.x * 16;
  const int r    = lane & 15;
  const int half = lane >> 4;          // 0/1
  const int row  = row0 + r;
  const int col  = ct * 16 + r;

  v8f acc = {};
#pragma unroll
  for (int ko = 0; ko < HD; ko += 32) {
    v16h a, b;
    const float* xp = X + (size_t)row * HD + ko + half * 16;
#pragma unroll
    for (int j = 0; j < 16; ++j) {
      a[j]     = (_Float16)xp[j];        // K = ko + half*16 + j
    }
    const float* wp = W + (size_t)(ko + half * 16) * HD + col;
#pragma unroll
    for (int j = 0; j < 16; ++j)
      b[j] = (_Float16)wp[(size_t)j * HD];  // K = ko + half*16 + j, N = col
    acc = __builtin_amdgcn_wmma_f32_16x16x32_f16(
        false, a, false, b, (short)0, acc, false, false);
  }
#pragma unroll
  for (int v = 0; v < 8; ++v)
    Hout[(size_t)(row0 + v + 8 * half) * HD + ct * 16 + r] = acc[v];
}

// wa[i] = sum_j W[i,j] * a[j]   (folds W_dst @ a_dst; 64 threads, one block)
__global__ void wa_kernel(const float* __restrict__ W,
                          const float* __restrict__ a,
                          float* __restrict__ wa)
{
  int i = threadIdx.x;
  float s = 0.f;
#pragma unroll 8
  for (int j = 0; j < HD; ++j) s += W[(size_t)i * HD + j] * a[j];
  wa[i] = s;
}

// out[n] = dot(X[n,:], vec)  — one wave32 per node, 2 features/lane.
__global__ __launch_bounds__(256) void dot64(const float* __restrict__ X,
                                             const float* __restrict__ vec,
                                             float* __restrict__ out, int N)
{
  int node = blockIdx.x * (blockDim.x >> 5) + (threadIdx.x >> 5);
  int lane = threadIdx.x & 31;
  if (node >= N) return;
  const float* xp = X + (size_t)node * HD;
  float s = xp[lane] * vec[lane] + xp[lane + 32] * vec[lane + 32];
#pragma unroll
  for (int off = 16; off > 0; off >>= 1) s += __shfl_down(s, off, 32);
  if (lane == 0) out[node] = s;
}

__global__ void fillk(float* __restrict__ p, float v, size_t n)
{
  size_t i = (size_t)blockIdx.x * blockDim.x + threadIdx.x;
  if (i < n) p[i] = v;
}

__global__ void fill_int(int* __restrict__ p, int v, int n)
{
  int i = blockIdx.x * blockDim.x + threadIdx.x;
  if (i < n) p[i] = v;
}

__global__ void copy_int(const int* __restrict__ s, int* __restrict__ d, int n)
{
  int i = blockIdx.x * blockDim.x + threadIdx.x;
  if (i < n) d[i] = s[i];
}

// ---- CSR build (done once; edge structure is identical for both layers) ----
__global__ __launch_bounds__(256) void count_dst(const int* __restrict__ dst,
                                                 int* __restrict__ count, int E)
{
  int i = blockIdx.x * blockDim.x + threadIdx.x;
  if (i < E) atomicAdd(&count[dst[i]], 1);
}

// single-workgroup chunked inclusive scan (LDS Hillis-Steele + carry)
__global__ __launch_bounds__(256) void scan_rowptr(const int* __restrict__ count,
                                                   int* __restrict__ rowptr, int n)
{
  __shared__ int buf[256];
  __shared__ int carry;
  if (threadIdx.x == 0) { carry = 0; rowptr[0] = 0; }
  __syncthreads();
  for (int base = 0; base < n; base += 256) {
    int i = base + (int)threadIdx.x;
    buf[threadIdx.x] = (i < n) ? count[i] : 0;
    __syncthreads();
#pragma unroll
    for (int off = 1; off < 256; off <<= 1) {
      int t = (threadIdx.x >= (unsigned)off) ? buf[threadIdx.x - off] : 0;
      __syncthreads();
      buf[threadIdx.x] += t;
      __syncthreads();
    }
    if (i < n) rowptr[i + 1] = carry + buf[threadIdx.x];
    __syncthreads();
    if (threadIdx.x == 0) carry += buf[255];
    __syncthreads();
  }
}

__global__ __launch_bounds__(256) void reorder_edges(
    const int* __restrict__ src, const int* __restrict__ dst,
    int* __restrict__ cursor, int* __restrict__ ssrc, int E)
{
  int i = blockIdx.x * blockDim.x + threadIdx.x;
  if (i >= E) return;
  int p = atomicAdd(&cursor[dst[i]], 1);
  ssrc[p] = src[i];
}

// ---------------------------------------------------------------------------
// Fused GAT aggregation, gather-side, zero float atomics.
// One wave32 per destination node; 2 features/lane (64 total). Online softmax
// (running max + rescale) over the node's incoming-edge list; h[src] rows are
// gathered as two coalesced 128B loads per edge (L2-resident).
// acc is accumulated read-modify-write (convs into the same dst run serially).
// ---------------------------------------------------------------------------
__global__ __launch_bounds__(256) void gat_aggregate(
    const int* __restrict__ rowptr, const int* __restrict__ ssrc,
    const float* __restrict__ als, const float* __restrict__ ald,
    const float* __restrict__ h, float* __restrict__ acc, int n_dst)
{
  int d    = blockIdx.x * (blockDim.x >> 5) + (threadIdx.x >> 5);
  int lane = threadIdx.x & 31;
  if (d >= n_dst) return;
  int beg = rowptr[d], end = rowptr[d + 1];
  if (beg == end) return;                       // no incoming edges: acc + bias only
  float aldd = ald[d];
  float m = -INFINITY, s = 0.f, a0 = 0.f, a1 = 0.f;
  for (int j = beg; j < end; ++j) {
    int sn = __builtin_amdgcn_readfirstlane(ssrc[j]);   // wave-uniform -> SGPR
    float el = als[sn] + aldd;
    el = el > 0.f ? el : 0.2f * el;                     // leaky_relu(0.2)
    float mn = fmaxf(m, el);
    float scale = __expf(m - mn);                       // first iter: exp(-inf)=0
    float w     = __expf(el - mn);
    s  = s * scale + w;
    const float* hp = h + (size_t)sn * HD;
    a0 = a0 * scale + w * hp[lane];
    a1 = a1 * scale + w * hp[lane + 32];
    m = mn;
  }
  float inv = 1.f / s;
  float* op = acc + (size_t)d * HD;
  op[lane]      += a0 * inv;
  op[lane + 32] += a1 * inv;
}

// x = relu(x + b1[f] (+ b2[f]))   per-feature biases from the two convs
__global__ void relu_bias2(float* __restrict__ x, const float* __restrict__ b1,
                           const float* __restrict__ b2, size_t n)
{
  size_t i = (size_t)blockIdx.x * blockDim.x + threadIdx.x;
  if (i >= n) return;
  int f = (int)(i & 63);
  float v = x[i] + b1[f] + (b2 ? b2[f] : 0.f);
  x[i] = v > 0.f ? v : 0.f;
}

// out[n,c] = dot(x[n,:], W[:,c]) + b[c]  (C = 4 or 2; thread per output)
__global__ void out_linear(const float* __restrict__ x,
                           const float* __restrict__ W,
                           const float* __restrict__ b,
                           float* __restrict__ out, int N, int C)
{
  size_t i = (size_t)blockIdx.x * blockDim.x + threadIdx.x;
  if (i >= (size_t)N * C) return;
  int n = (int)(i / C), c = (int)(i % C);
  float s = b[c];
#pragma unroll 8
  for (int f = 0; f < HD; ++f) s += x[(size_t)n * HD + f] * W[f * C + c];
  out[i] = s;
}

// ---------------------------------------------------------------------------
static void build_csr(const int* ei, int E, int n_dst, int* ssrc, int* rowptr,
                      int* tmp, hipStream_t stream)
{
  const int* src = ei;
  const int* dst = ei + E;
  fill_int<<<(n_dst + 255) / 256, 256, 0, stream>>>(tmp, 0, n_dst);
  count_dst<<<(E + 255) / 256, 256, 0, stream>>>(dst, tmp, E);
  scan_rowptr<<<1, 256, 0, stream>>>(tmp, rowptr, n_dst);
  copy_int<<<(n_dst + 255) / 256, 256, 0, stream>>>(rowptr, tmp, n_dst);
  reorder_edges<<<(E + 255) / 256, 256, 0, stream>>>(src, dst, tmp, ssrc, E);
}

static void run_conv(const float* xs, const float* xd, int n_src, int n_dst,
                     const int* rowptr, const int* ssrc,
                     const float* Wsrc, const float* Wdst,
                     const float* asrc, const float* adst,
                     float* h, float* als, float* ald, float* wa,
                     float* acc, hipStream_t stream)
{
  wa_kernel<<<1, 64, 0, stream>>>(Wdst, adst, wa);
  dim3 gblk(32, 4);
  gemm64_wmma<<<n_src / 16, gblk, 0, stream>>>(xs, Wsrc, h);
  dot64<<<(n_src + 7) / 8, 256, 0, stream>>>(h, asrc, als, n_src);
  dot64<<<(n_dst + 7) / 8, 256, 0, stream>>>(xd, wa, ald, n_dst);
  gat_aggregate<<<(n_dst + 7) / 8, 256, 0, stream>>>(rowptr, ssrc, als, ald,
                                                     h, acc, n_dst);
}

extern "C" void kernel_launch(void* const* d_in, const int* in_sizes, int n_in,
                              void* d_out, int out_size, void* d_ws, size_t ws_size,
                              hipStream_t stream)
{
  const float* x_bus  = (const float*)d_in[0];
  const float* x_gen  = (const float*)d_in[1];
  const int*   ei_bb  = (const int*)d_in[2];
  const int*   ei_gb  = (const int*)d_in[3];
  const int*   ei_bg  = (const int*)d_in[4];
  const float* Ws     = (const float*)d_in[5];   // [2,3,2,64,64]
  const float* atts   = (const float*)d_in[6];   // [2,3,2,64]
  const float* biases = (const float*)d_in[7];   // [2,3,64]
  const float* lWb    = (const float*)d_in[8];
  const float* lbb    = (const float*)d_in[9];
  const float* lWg    = (const float*)d_in[10];
  const float* lbg    = (const float*)d_in[11];
  float* out = (float*)d_out;

  float* ws = (float*)d_ws;
  const size_t F = (size_t)NB * HD;
  float* bufB0 = ws;             // layer-0 bus output / layer-1 bus input
  float* bufG0 = bufB0 + F;      // layer-0 gen output / layer-1 gen input
  float* bufB1 = bufG0 + F;      // layer-1 bus output
  float* bufG1 = bufB1 + F;      // layer-1 gen output
  float* h     = bufG1 + F;      // shared h_src buffer (one conv at a time)
  float* als   = h + F;
  float* ald   = als + NB;
  float* wa    = ald + NB;       // 64 floats
  int* sorted_bb = (int*)(wa + 64);
  int* sorted_gb = sorted_bb + EBB;
  int* sorted_bg = sorted_gb + EGB;
  int* rp_bb     = sorted_bg + EBG;   // NB+1
  int* rp_gb     = rp_bb + NB + 1;
  int* rp_bg     = rp_gb + NB + 1;
  int* tmp       = rp_bg + NG + 1;    // NB ints scratch

  // CSR by destination, built once per call, reused by both layers
  build_csr(ei_bb, EBB, NB, sorted_bb, rp_bb, tmp, stream);
  build_csr(ei_gb, EGB, NB, sorted_gb, rp_gb, tmp, stream);
  build_csr(ei_bg, EBG, NG, sorted_bg, rp_bg, tmp, stream);

  for (int l = 0; l < 2; ++l) {
    const float* xinB = (l == 0) ? x_bus : bufB0;
    const float* xinG = (l == 0) ? x_gen : bufG0;
    float* outB = (l == 0) ? bufB0 : bufB1;
    float* outG = (l == 0) ? bufG0 : bufG1;

    fillk<<<(unsigned)((F + 255) / 256), 256, 0, stream>>>(outB, 0.f, F);
    fillk<<<(unsigned)((F + 255) / 256), 256, 0, stream>>>(outG, 0.f, F);

    for (int t = 0; t < 3; ++t) {
      const float* Wsrc = Ws   + (size_t)((l * 3 + t) * 2 + 0) * 4096;
      const float* Wdst = Ws   + (size_t)((l * 3 + t) * 2 + 1) * 4096;
      const float* asrc = atts + (size_t)((l * 3 + t) * 2 + 0) * 64;
      const float* adst = atts + (size_t)((l * 3 + t) * 2 + 1) * 64;
      if (t == 0)       // bus -> bus
        run_conv(xinB, xinB, NB, NB, rp_bb, sorted_bb, Wsrc, Wdst, asrc, adst,
                 h, als, ald, wa, outB, stream);
      else if (t == 1)  // gen -> bus
        run_conv(xinG, xinB, NG, NB, rp_gb, sorted_gb, Wsrc, Wdst, asrc, adst,
                 h, als, ald, wa, outB, stream);
      else              // bus -> gen
        run_conv(xinB, xinG, NB, NG, rp_bg, sorted_bg, Wsrc, Wdst, asrc, adst,
                 h, als, ald, wa, outG, stream);
    }
    relu_bias2<<<(unsigned)((F + 255) / 256), 256, 0, stream>>>(
        outB, biases + (size_t)(l * 3 + 0) * 64, biases + (size_t)(l * 3 + 1) * 64, F);
    relu_bias2<<<(unsigned)((F + 255) / 256), 256, 0, stream>>>(
        outG, biases + (size_t)(l * 3 + 2) * 64, nullptr, F);
  }

  out_linear<<<(NB * 4 + 255) / 256, 256, 0, stream>>>(bufB1, lWb, lbb, out, NB, 4);
  out_linear<<<(NG * 2 + 255) / 256, 256, 0, stream>>>(bufG1, lWg, lbg,
                                                       out + (size_t)NB * 4, NG, 2);
}
